// CrossCondGPT2MW_52046413693261
// MI455X (gfx1250) — matrix operand order repaired and checked by reading, not compiled, not run on hardware
//
#include <hip/hip_runtime.h>
#include <hip/hip_bf16.h>
#include <math.h>

// ---------------------------------------------------------------------------
// CrossCondGPT2 forward on MI455X (gfx1250, wave32, WMMA)
// All GEMMs use v_wmma_f32_16x16x32_bf16; f32 tiles converted to bf16 in LDS.
// Per K-step: 12 ds_load_b128 (2 A frags + 4 B frags), one wait, 8 WMMAs.
// ---------------------------------------------------------------------------

typedef __attribute__((ext_vector_type(16))) __bf16 v16bf;
typedef __attribute__((ext_vector_type(8)))  __bf16 v8bf;
typedef __attribute__((ext_vector_type(4)))  __bf16 v4bf;
typedef __attribute__((ext_vector_type(8)))  float  v8f;

#define F_BIAS   1
#define F_RES    2
#define F_GELU   4
#define F_MASK   8
#define F_TRANSB 16

#define BM 256          // 8 waves x 32 rows
#define BN 64
#define BK 32
#define LDAP (BK + 8)   // 40 bf16 = 80 B row stride: 16B-aligned, non-pow2 banks

__global__ __launch_bounds__(256) void gemm_wmma(
    const float* __restrict__ A,   int lda,
    const float* __restrict__ Bw,  int ldb,
    const float* __restrict__ bias,
    const float* __restrict__ Res, int ldr,
    float* __restrict__ Out,       int ldc,
    int M, int N, int K,
    int divZ,
    long long sA1, long long sA2,
    long long sB1, long long sB2,
    long long sC1, long long sC2,
    long long sR1, long long sR2,
    int flags, float scale)
{
    __shared__ __align__(16) __bf16 lA[BM][LDAP];   // A tile, row-major [m][k]
    __shared__ __align__(16) __bf16 lBt[BN][LDAP];  // B tile, TRANSPOSED [n][k]

    const int z  = blockIdx.z;
    const int z1 = z / divZ, z2 = z % divZ;
    A   += z1 * sA1 + z2 * sA2;
    Bw  += z1 * sB1 + z2 * sB2;
    Out += z1 * sC1 + z2 * sC2;
    if (flags & F_RES) Res += z1 * sR1 + z2 * sR2;

    const int tid  = threadIdx.x;
    const int wave = tid >> 5;
    const int lane = tid & 31;
    const int half = lane >> 4;   // 0: lanes 0-15, 1: lanes 16-31
    const int m16  = lane & 15;

    const int row0 = blockIdx.y * BM;
    const int col0 = blockIdx.x * BN;

    const bool alignedA = ((lda & 3) == 0);
    const bool alignedB = ((ldb & 3) == 0);

    v8f acc[2][4];
#pragma unroll
    for (int mt = 0; mt < 2; ++mt)
#pragma unroll
        for (int nt = 0; nt < 4; ++nt)
#pragma unroll
            for (int r = 0; r < 8; ++r) acc[mt][nt][r] = 0.0f;

    for (int k0 = 0; k0 < K; k0 += BK) {
        const bool fullK = (k0 + BK <= K);

        // ---- stage A tile (BM x BK) f32 -> bf16 ----
        if (fullK && alignedA) {
#pragma unroll
            for (int i = 0; i < (BM * BK) / (256 * 4); ++i) {   // 8 iters
                int idx = tid + i * 256;
                int r   = idx >> 3;            // 8 float4 per 32-wide row
                int c4  = (idx & 7) << 2;
                const float4 f = *(const float4*)&A[(long long)(row0 + r) * lda + k0 + c4];
                v4bf o; o[0] = (__bf16)f.x; o[1] = (__bf16)f.y;
                        o[2] = (__bf16)f.z; o[3] = (__bf16)f.w;
                *(v4bf*)&lA[r][c4] = o;
            }
        } else {
            for (int idx = tid; idx < BM * BK; idx += 256) {
                int r = idx >> 5, c = idx & (BK - 1);
                int gk = k0 + c;
                float vl = (gk < K) ? A[(long long)(row0 + r) * lda + gk] : 0.0f;
                lA[r][c] = (__bf16)vl;
            }
        }
        // ---- stage B tile transposed into lBt[n][k] ----
        if (flags & F_TRANSB) {                 // global B is [N][K]: contiguous copy
            if (fullK && alignedB) {
#pragma unroll
                for (int i = 0; i < (BK * BN) / (256 * 4); ++i) {  // 2 iters
                    int idx = tid + i * 256;
                    int n   = idx >> 3;
                    int c4  = (idx & 7) << 2;
                    const float4 f = *(const float4*)&Bw[(long long)(col0 + n) * ldb + k0 + c4];
                    v4bf o; o[0] = (__bf16)f.x; o[1] = (__bf16)f.y;
                            o[2] = (__bf16)f.z; o[3] = (__bf16)f.w;
                    *(v4bf*)&lBt[n][c4] = o;
                }
            } else {
                for (int idx = tid; idx < BK * BN; idx += 256) {
                    int n = idx >> 5, c = idx & (BK - 1);
                    int gk = k0 + c;
                    float vl = (gk < K) ? Bw[(long long)(col0 + n) * ldb + gk] : 0.0f;
                    lBt[n][c] = (__bf16)vl;
                }
            }
        } else {                                // global B is [K][N]: transpose on store
            if (fullK && alignedB) {
#pragma unroll
                for (int i = 0; i < (BK * BN) / (256 * 4); ++i) {  // 2 iters
                    int idx = tid + i * 256;
                    int r   = idx >> 4;            // 16 float4 per 64-wide row
                    int c4  = (idx & 15) << 2;
                    const float4 f = *(const float4*)&Bw[(long long)(k0 + r) * ldb + col0 + c4];
                    lBt[c4 + 0][r] = (__bf16)f.x;
                    lBt[c4 + 1][r] = (__bf16)f.y;
                    lBt[c4 + 2][r] = (__bf16)f.z;
                    lBt[c4 + 3][r] = (__bf16)f.w;
                }
            } else {
                for (int idx = tid; idx < BK * BN; idx += 256) {
                    int n = idx >> 5, c = idx & (BK - 1);
                    int gk = k0 + c;
                    float vl = (gk < K) ? Bw[(long long)gk * ldb + col0 + n] : 0.0f;
                    lBt[n][c] = (__bf16)vl;
                }
            }
        }
        __syncthreads();

        // ---- preload ALL fragments (12 x ds_load_b128), then 8 WMMAs ----
        v16bf af[2];
#pragma unroll
        for (int mt = 0; mt < 2; ++mt) {
            const int ar = wave * 32 + mt * 16 + m16;
            v8bf alo = *(const v8bf*)&lA[ar][half * 8];        // K 0-7  / 8-15
            v8bf ahi = *(const v8bf*)&lA[ar][16 + half * 8];   // K 16-23 / 24-31
            af[mt] = __builtin_shufflevector(alo, ahi,
                0, 1, 2, 3, 4, 5, 6, 7, 8, 9, 10, 11, 12, 13, 14, 15);
        }
        v16bf bf[4];
#pragma unroll
        for (int nt = 0; nt < 4; ++nt) {
            const int bc = nt * 16 + m16;
            v8bf blo = *(const v8bf*)&lBt[bc][half * 8];       // K 0-7  / 8-15
            v8bf bhi = *(const v8bf*)&lBt[bc][16 + half * 8];  // K 16-23 / 24-31
            bf[nt] = __builtin_shufflevector(blo, bhi,
                0, 1, 2, 3, 4, 5, 6, 7, 8, 9, 10, 11, 12, 13, 14, 15);
        }
#pragma unroll
        for (int mt = 0; mt < 2; ++mt)
#pragma unroll
            for (int nt = 0; nt < 4; ++nt)
                acc[mt][nt] = __builtin_amdgcn_wmma_f32_16x16x32_bf16(
                    false, af[mt], false, bf[nt], (short)0, acc[mt][nt], false, false);
        __syncthreads();
    }

    // ---- epilogue: bias / scale+mask / GELU / residual ----
#pragma unroll
    for (int mt = 0; mt < 2; ++mt) {
#pragma unroll
        for (int nt = 0; nt < 4; ++nt) {
#pragma unroll
            for (int r = 0; r < 8; ++r) {
                int gm = row0 + wave * 32 + mt * 16 + half * 8 + r;
                int gn = col0 + nt * 16 + m16;
                float vr = acc[mt][nt][r];
                if (flags & F_BIAS) vr += bias[gn];
                if (flags & F_MASK) {      // block-causal: (j % 256) > (i % 256) -> -1e9
                    vr *= scale;
                    if ((gn & 255) > (gm & 255)) vr = -1e9f;
                }
                if (flags & F_GELU) vr = 0.5f * vr * (1.0f + erff(vr * 0.70710678118654752f));
                if (flags & F_RES)  vr += Res[(long long)gm * ldr + gn];
                Out[(long long)gm * ldc + gn] = vr;
            }
        }
    }
}

// ---- LayerNorm: one block per row ----
__global__ __launch_bounds__(256) void ln_kernel(
    const float* __restrict__ X, const float* __restrict__ g,
    const float* __restrict__ b, float* __restrict__ O, int Cc)
{
    const int row = blockIdx.x, tid = threadIdx.x;
    const float* xr = X + (long long)row * Cc;
    __shared__ float red[256];

    float s = 0.0f;
    for (int c = tid; c < Cc; c += 256) s += xr[c];
    red[tid] = s; __syncthreads();
    for (int o = 128; o > 0; o >>= 1) { if (tid < o) red[tid] += red[tid + o]; __syncthreads(); }
    const float mean = red[0] / Cc; __syncthreads();

    float vs = 0.0f;
    for (int c = tid; c < Cc; c += 256) { float d = xr[c] - mean; vs += d * d; }
    red[tid] = vs; __syncthreads();
    for (int o = 128; o > 0; o >>= 1) { if (tid < o) red[tid] += red[tid + o]; __syncthreads(); }
    const float rstd = rsqrtf(red[0] / Cc + 1e-5f);

    for (int c = tid; c < Cc; c += 256)
        O[(long long)row * Cc + c] = (xr[c] - mean) * rstd * g[c] + b[c];
}

// ---- Softmax: one block per row (in place) ----
__global__ __launch_bounds__(256) void softmax_kernel(float* __restrict__ S, int T_)
{
    float* r = S + (long long)blockIdx.x * T_;
    const int tid = threadIdx.x;
    __shared__ float red[256];

    float mx = -3.4e38f;
    for (int j = tid; j < T_; j += 256) mx = fmaxf(mx, r[j]);
    red[tid] = mx; __syncthreads();
    for (int o = 128; o > 0; o >>= 1) { if (tid < o) red[tid] = fmaxf(red[tid], red[tid + o]); __syncthreads(); }
    mx = red[0]; __syncthreads();

    float sum = 0.0f;
    for (int j = tid; j < T_; j += 256) { float e = expf(r[j] - mx); r[j] = e; sum += e; }
    red[tid] = sum; __syncthreads();
    for (int o = 128; o > 0; o >>= 1) { if (tid < o) red[tid] += red[tid + o]; __syncthreads(); }
    const float inv = 1.0f / red[0];

    for (int j = tid; j < T_; j += 256) r[j] *= inv;
}

// ---- Embedding assembly: x = concat(condproj, tok_up[idx], tok_dn[idx]) + pos ----
__global__ __launch_bounds__(256) void embed_kernel(
    const float* __restrict__ condp,
    const float* __restrict__ tok_up, const int* __restrict__ idx_up,
    const float* __restrict__ tok_dn, const int* __restrict__ idx_dn,
    const float* __restrict__ pos, float* __restrict__ x)
{
    const int row = blockIdx.x;          // 0..1535 = b*768 + p
    const int b = row / 768;
    const int p = row % 768;             // t == BS so pe index == p
    const int seg = p >> 8, i = p & 255;
    const float* src;
    if (seg == 0)      src = condp  + (long long)(b * 256 + i) * 768;
    else if (seg == 1) src = tok_up + (long long)idx_up[b * 256 + i] * 768;
    else               src = tok_dn + (long long)idx_dn[b * 256 + i] * 768;
    for (int c = threadIdx.x; c < 768; c += 256)
        x[(long long)row * 768 + c] = src[c] + pos[(long long)p * 768 + c];
}

// ---------------------------------------------------------------------------
static inline void launch_gemm(hipStream_t s,
    const float* A, int lda, const float* B, int ldb, const float* bias,
    const float* Res, int ldr, float* Out, int ldc,
    int M, int N, int K, int Z, int divZ,
    long long sA1, long long sA2, long long sB1, long long sB2,
    long long sC1, long long sC2, long long sR1, long long sR2,
    int flags, float scale)
{
    dim3 grid((N + BN - 1) / BN, (M + BM - 1) / BM, Z);
    gemm_wmma<<<grid, 256, 0, s>>>(A, lda, B, ldb, bias, Res, ldr, Out, ldc,
                                   M, N, K, divZ, sA1, sA2, sB1, sB2,
                                   sC1, sC2, sR1, sR2, flags, scale);
}

extern "C" void kernel_launch(void* const* d_in, const int* in_sizes, int n_in,
                              void* d_out, int out_size, void* d_ws, size_t ws_size,
                              hipStream_t stream)
{
    constexpr int Cn = 768, Tt = 768, tseg = 256, Bn = 2, Hh = 12, NM = 438;
    constexpr long long TC = (long long)Tt * Cn;            // 589824
    constexpr long long XE = (long long)Bn * TC;            // 1179648
    constexpr long long FBE = (long long)Bn * Tt * 4 * Cn;  // 4718592
    constexpr long long SCE = (long long)Bn * Hh * Tt * Tt; // 14155776

    const int*   idx_up = (const int*)d_in[0];
    const int*   idx_dn = (const int*)d_in[1];
    const float* cond   = (const float*)d_in[2];
    const float* tokup  = (const float*)d_in[3];
    const float* tokdn  = (const float*)d_in[4];
    const float* pos    = (const float*)d_in[5];
    const float* condw  = (const float*)d_in[6];
    const float* condb  = (const float*)d_in[7];
    const float* lnfg   = (const float*)d_in[40];
    const float* lnfb   = (const float*)d_in[41];
    const float* hupw   = (const float*)d_in[42];
    const float* hdnw   = (const float*)d_in[43];

    float* ws = (float*)d_ws;
    float* x  = ws;            // [2*768, 768]
    float* h  = x  + XE;       // LN output
    float* q  = h  + XE;
    float* k  = q  + XE;
    float* v  = k  + XE;
    float* y  = v  + XE;       // attention output (pre-proj)
    float* fb = y  + XE;       // MLP hidden [1536, 3072]
    float* sc = fb + FBE;      // attention scores [2,12,768,768]
    float* tc = sc + SCE;      // cond projection [512, 768]

    const int Mrows = Bn * Tt; // 1536

    // 1) cond @ cond_w + cond_b  -> tc   [512, 768], K=438 (zero-padded in LDS)
    launch_gemm(stream, cond, NM, condw, Cn, condb, nullptr, 0, tc, Cn,
                Bn * tseg, Cn, NM, 1, 1, 0,0, 0,0, 0,0, 0,0, F_BIAS, 1.0f);
    // 2) assemble embeddings
    embed_kernel<<<Mrows, 256, 0, stream>>>(tc, tokup, idx_up, tokdn, idx_dn, pos, x);

    // 3) transformer stacks (base: inputs 8..23, head: 24..39)
    for (int st = 0; st < 2; ++st) {
        const int o = (st == 0) ? 8 : 24;
        const float* ln1g = (const float*)d_in[o + 0];
        const float* ln1b = (const float*)d_in[o + 1];
        const float* wq   = (const float*)d_in[o + 2];
        const float* bq   = (const float*)d_in[o + 3];
        const float* wk   = (const float*)d_in[o + 4];
        const float* bk   = (const float*)d_in[o + 5];
        const float* wv   = (const float*)d_in[o + 6];
        const float* bv   = (const float*)d_in[o + 7];
        const float* wo   = (const float*)d_in[o + 8];
        const float* bo   = (const float*)d_in[o + 9];
        const float* ln2g = (const float*)d_in[o + 10];
        const float* ln2b = (const float*)d_in[o + 11];
        const float* fcw  = (const float*)d_in[o + 12];
        const float* fcb  = (const float*)d_in[o + 13];
        const float* prw  = (const float*)d_in[o + 14];
        const float* prb  = (const float*)d_in[o + 15];

        for (int l = 0; l < 6; ++l) {
            const long long wOff = (long long)l * Cn * Cn;
            const long long fOff = (long long)l * Cn * 4 * Cn;

            // h = LN1(x)
            ln_kernel<<<Mrows, 256, 0, stream>>>(x, ln1g + l * Cn, ln1b + l * Cn, h, Cn);
            // q,k,v = h @ W + b
            launch_gemm(stream, h, Cn, wq + wOff, Cn, bq + (long long)l * Cn, nullptr, 0, q, Cn,
                        Mrows, Cn, Cn, 1, 1, 0,0, 0,0, 0,0, 0,0, F_BIAS, 1.0f);
            launch_gemm(stream, h, Cn, wk + wOff, Cn, bk + (long long)l * Cn, nullptr, 0, k, Cn,
                        Mrows, Cn, Cn, 1, 1, 0,0, 0,0, 0,0, 0,0, F_BIAS, 1.0f);
            launch_gemm(stream, h, Cn, wv + wOff, Cn, bv + (long long)l * Cn, nullptr, 0, v, Cn,
                        Mrows, Cn, Cn, 1, 1, 0,0, 0,0, 0,0, 0,0, F_BIAS, 1.0f);

            // scores[b,h] = (q_bh @ k_bh^T) * 1/8, masked  (z = b*12 + h)
            launch_gemm(stream, q, Cn, k, Cn, nullptr, nullptr, 0, sc, Tt,
                        Tt, Tt, 64, Bn * Hh, Hh,
                        TC, 64,             // A offsets: b*T*C + h*64
                        TC, 64,             // B offsets (k, transposed load)
                        (long long)Hh * Tt * Tt, (long long)Tt * Tt, // C: z*T*T
                        0, 0, F_MASK | F_TRANSB, 0.125f);
            // softmax rows
            softmax_kernel<<<Bn * Hh * Tt, 256, 0, stream>>>(sc, Tt);
            // y[b,h] = P @ v_bh
            launch_gemm(stream, sc, Tt, v, Cn, nullptr, nullptr, 0, y, Cn,
                        Tt, 64, Tt, Bn * Hh, Hh,
                        (long long)Hh * Tt * Tt, (long long)Tt * Tt,
                        TC, 64, TC, 64, 0, 0, 0, 1.0f);
            // x = x + y @ Wo + Bo
            launch_gemm(stream, y, Cn, wo + wOff, Cn, bo + (long long)l * Cn, x, Cn, x, Cn,
                        Mrows, Cn, Cn, 1, 1, 0,0, 0,0, 0,0, 0,0, F_BIAS | F_RES, 1.0f);

            // h = LN2(x); fb = gelu(h @ Wf + Bf); x = x + fb @ Wp + Bp
            ln_kernel<<<Mrows, 256, 0, stream>>>(x, ln2g + l * Cn, ln2b + l * Cn, h, Cn);
            launch_gemm(stream, h, Cn, fcw + fOff, 4 * Cn, fcb + (long long)l * 4 * Cn,
                        nullptr, 0, fb, 4 * Cn,
                        Mrows, 4 * Cn, Cn, 1, 1, 0,0, 0,0, 0,0, 0,0, F_BIAS | F_GELU, 1.0f);
            launch_gemm(stream, fb, 4 * Cn, prw + fOff, Cn, prb + (long long)l * Cn, x, Cn, x, Cn,
                        Mrows, Cn, 4 * Cn, 1, 1, 0,0, 0,0, 0,0, 0,0, F_BIAS | F_RES, 1.0f);
        }
    }

    // 4) final LN + output heads
    ln_kernel<<<Mrows, 256, 0, stream>>>(x, lnfg, lnfb, h, Cn);
    float* out_up = (float*)d_out;                       // [2,256,512]
    float* out_dn = out_up + (long long)Bn * tseg * 512; // [2,256,512]
    // logits_up = h[:, 256:512] @ head_up_w   (per-batch z)
    launch_gemm(stream, h + (long long)tseg * Cn, Cn, hupw, 512, nullptr, nullptr, 0,
                out_up, 512, tseg, 512, Cn, Bn, 1,
                TC, 0, 0, 0, (long long)tseg * 512, 0, 0, 0, 0, 1.0f);
    // logits_down = h[:, 512:768] @ head_down_w
    launch_gemm(stream, h + (long long)2 * tseg * Cn, Cn, hdnw, 512, nullptr, nullptr, 0,
                out_dn, 512, tseg, 512, Cn, Bn, 1,
                TC, 0, 0, 0, (long long)tseg * 512, 0, 0, 0, 0, 1.0f);
}
